// ModifiedTransformer_60112362275491
// MI455X (gfx1250) — matrix operand order repaired and checked by reading, not compiled, not run on hardware
//
#include <hip/hip_runtime.h>
#include <hip/hip_bf16.h>
#include <math.h>

// ---------------- problem constants ----------------
#define BB      8
#define LLEN    1024
#define IN_F    32
#define DMODEL  512
#define NH      8
#define HD      64          // DMODEL / NH
#define NLAYERS 3
#define DFFN    2048
#define SKEYS   35          // SAMPLE_K
#define NTOP    35
#define MROWS   (BB * LLEN) // 8192
#define LN_EPS  1e-5f

typedef __bf16 bf16_t;
typedef __attribute__((ext_vector_type(16))) __bf16       v16bf;
typedef __attribute__((ext_vector_type(8)))  float        v8f;
typedef __attribute__((ext_vector_type(4)))  unsigned int u32x4;

// ---------------- WMMA helpers (CDNA5 wave32 layouts) ----------------
__device__ __forceinline__ v8f wmma_bf16(v16bf a, v16bf b, v8f c) {
  return __builtin_amdgcn_wmma_f32_16x16x32_bf16(false, a, false, b, (short)0, c,
                                                 false, false);
}

// A-matrix 16x32 bf16 fragment, rows r0..r0+15, K chunk k0..k0+31.
// lanes 0-15: M=lane,   K = k0+0..7  (v0-3) and k0+16..23 (v4-7)
// lanes16-31: M=lane-16,K = k0+8..15 (v0-3) and k0+24..31 (v4-7)
__device__ __forceinline__ v16bf load_frag_row(const bf16_t* base, int ld, int r0,
                                               int k0, int lane) {
  int r  = r0 + (lane & 15);
  int kb = k0 + ((lane & 16) ? 8 : 0);
  const bf16_t* p = base + (size_t)r * ld + kb;
  union { v16bf v; u32x4 q[2]; } t;
  t.q[0] = *(const u32x4*)(p);
  t.q[1] = *(const u32x4*)(p + 16);
  return t.v;
}

// B-matrix 32x16 bf16 fragment from a (N,K) row-major array:
// lanes 0-15: N=lane,   K = k0+0..7  (v0-3) and k0+16..23 (v4-7)
// lanes16-31: N=lane-16,K = k0+8..15 (v0-3) and k0+24..31 (v4-7)
__device__ __forceinline__ v16bf load_frag_col(const bf16_t* wt, int ld, int n0,
                                               int k0, int lane) {
  int n  = n0 + (lane & 15);
  int kb = k0 + ((lane & 16) ? 8 : 0);
  const bf16_t* p = wt + (size_t)n * ld + kb;
  union { v16bf v; u32x4 q[2]; } t;
  t.q[0] = *(const u32x4*)(p);
  t.q[1] = *(const u32x4*)(p + 16);
  return t.v;
}

// ---------------- weight convert + transpose: W(K,N) f32 -> WT(N,K) bf16 -------
__global__ void k_wt(const float* __restrict__ W, bf16_t* __restrict__ WT, int K,
                     int N) {
  size_t t = (size_t)blockIdx.x * blockDim.x + threadIdx.x;
  if (t >= (size_t)K * N) return;
  int n = (int)(t % N);
  int k = (int)(t / N);
  WT[(size_t)n * K + k] = (bf16_t)W[t];
}

// ---------------- embedding: h = x@emb_w + emb_b + pe_w*pe ----------------
__global__ void k_embed(const float* __restrict__ x, const float* __restrict__ ew,
                        const float* __restrict__ eb, const float* __restrict__ pe,
                        const float* __restrict__ pew, float* __restrict__ hf,
                        bf16_t* __restrict__ hb) {
  int t = blockIdx.x * blockDim.x + threadIdx.x;
  if (t >= MROWS * DMODEL) return;
  int c = t % DMODEL;
  int row = t / DMODEL;
  int l = row % LLEN;
  const float* xr = x + (size_t)row * IN_F;
  float a = eb[c];
#pragma unroll
  for (int i = 0; i < IN_F; ++i) a += xr[i] * ew[i * DMODEL + c];
  a += pew[0] * pe[(size_t)l * DMODEL + c];
  hf[t] = a;
  hb[t] = (bf16_t)a;
}

// ---------------- WMMA GEMM: C[M,N] = act(A[M,K]@WT[N,K]^T + bias) -------------
// Register-blocked + software-pipelined:
//   block = 4 waves tiling along M; each wave owns 64(M) x 16(N):
//   4 accumulator tiles reuse one B fragment (weights) per K chunk, and the
//   fragments for chunk k+32 are loaded while the 4 WMMAs for chunk k execute.
__global__ __launch_bounds__(128) void k_gemm(const bf16_t* __restrict__ A,
                                              const bf16_t* __restrict__ WT,
                                              const float* __restrict__ bias,
                                              float* __restrict__ Cf,
                                              bf16_t* __restrict__ Cb, int N, int K,
                                              int act) {
  int lane = threadIdx.x & 31;
  int wave = threadIdx.x >> 5;
  int m0 = blockIdx.x * 256 + wave * 64;  // 4 x 16-row tiles per wave
  int n0 = blockIdx.y * 16;
  v8f acc[4];
#pragma unroll
  for (int t = 0; t < 4; ++t)
    acc[t] = (v8f){0.f, 0.f, 0.f, 0.f, 0.f, 0.f, 0.f, 0.f};

  v16bf a0 = load_frag_row(A, K, m0 +  0, 0, lane);
  v16bf a1 = load_frag_row(A, K, m0 + 16, 0, lane);
  v16bf a2 = load_frag_row(A, K, m0 + 32, 0, lane);
  v16bf a3 = load_frag_row(A, K, m0 + 48, 0, lane);
  v16bf b0 = load_frag_col(WT, K, n0, 0, lane);

  for (int k0 = 0; k0 < K; k0 += 32) {
    v16bf na0 = a0, na1 = a1, na2 = a2, na3 = a3, nb = b0;
    int kn = k0 + 32;
    if (kn < K) {  // issue next-chunk loads before consuming current fragments
      na0 = load_frag_row(A, K, m0 +  0, kn, lane);
      na1 = load_frag_row(A, K, m0 + 16, kn, lane);
      na2 = load_frag_row(A, K, m0 + 32, kn, lane);
      na3 = load_frag_row(A, K, m0 + 48, kn, lane);
      nb  = load_frag_col(WT, K, n0, kn, lane);
    }
    acc[0] = wmma_bf16(a0, b0, acc[0]);
    acc[1] = wmma_bf16(a1, b0, acc[1]);
    acc[2] = wmma_bf16(a2, b0, acc[2]);
    acc[3] = wmma_bf16(a3, b0, acc[3]);
    a0 = na0; a1 = na1; a2 = na2; a3 = na3; b0 = nb;
  }

  int n = n0 + (lane & 15);
  int rbase = (lane & 16) ? 8 : 0;
  float bv = bias ? bias[n] : 0.0f;
#pragma unroll
  for (int t = 0; t < 4; ++t)
#pragma unroll
    for (int r = 0; r < 8; ++r) {
      int m = m0 + t * 16 + rbase + r;
      float v = acc[t][r] + bv;
      if (act == 1) v = 0.5f * v * (1.0f + erff(v * 0.70710678118654752f));
      if (Cf) Cf[(size_t)m * N + n] = v;
      if (Cb) Cb[(size_t)m * N + n] = (bf16_t)v;
    }
}

// ---------------- flash full attention, one wave per (bh, 16-query tile) -------
__global__ __launch_bounds__(32) void k_flash(const bf16_t* __restrict__ Q,
                                              const bf16_t* __restrict__ Km,
                                              const bf16_t* __restrict__ Vm,
                                              bf16_t* __restrict__ O) {
  __shared__ bf16_t Pl[16 * 32];  // P tile (row-major, ld=32) for A fragments
  __shared__ bf16_t Vt[64 * 32];  // V chunk transposed: Vt[d][key]
  int lane = threadIdx.x;
  int bh = blockIdx.y;
  int b = bh / NH, h = bh % NH;
  int l0 = blockIdx.x * 16;
  const int ld = NH * HD;  // 512
  const bf16_t* Qp = Q + ((size_t)b * LLEN) * ld + h * HD;
  const bf16_t* Kp = Km + ((size_t)b * LLEN) * ld + h * HD;
  const bf16_t* Vp = Vm + ((size_t)b * LLEN) * ld + h * HD;

  v16bf qf0 = load_frag_row(Qp, ld, l0, 0, lane);
  v16bf qf1 = load_frag_row(Qp, ld, l0, 32, lane);

  v8f ctx[4];
  float rm[8], rs[8];
#pragma unroll
  for (int t = 0; t < 4; ++t) ctx[t] = (v8f){0.f, 0.f, 0.f, 0.f, 0.f, 0.f, 0.f, 0.f};
#pragma unroll
  for (int r = 0; r < 8; ++r) { rm[r] = -3.0e38f; rs[r] = 0.f; }

  const float scale = 0.125f;  // 1/sqrt(64)
  int col = lane & 15;
  int rbase = (lane & 16) ? 8 : 0;

  for (int s0 = 0; s0 < LLEN; s0 += 32) {
    // S = Q K^T for two 16-key tiles
    v8f S0 = (v8f){0.f, 0.f, 0.f, 0.f, 0.f, 0.f, 0.f, 0.f};
    v8f S1 = S0;
    {
      v16bf b00 = load_frag_col(Kp, ld, s0, 0, lane);
      v16bf b01 = load_frag_col(Kp, ld, s0, 32, lane);
      S0 = wmma_bf16(qf0, b00, S0);
      S0 = wmma_bf16(qf1, b01, S0);
      v16bf b10 = load_frag_col(Kp, ld, s0 + 16, 0, lane);
      v16bf b11 = load_frag_col(Kp, ld, s0 + 16, 32, lane);
      S1 = wmma_bf16(qf0, b10, S1);
      S1 = wmma_bf16(qf1, b11, S1);
    }
    float lm[8], p0[8], p1[8], ps[8], sc[8];
#pragma unroll
    for (int r = 0; r < 8; ++r) {
      float a = S0[r] * scale, c = S1[r] * scale;
      S0[r] = a; S1[r] = c;
      lm[r] = fmaxf(a, c);
    }
    // row max across the 16-lane half-group (wave32)
#pragma unroll
    for (int off = 1; off < 16; off <<= 1)
#pragma unroll
      for (int r = 0; r < 8; ++r) lm[r] = fmaxf(lm[r], __shfl_xor(lm[r], off, 32));
#pragma unroll
    for (int r = 0; r < 8; ++r) {
      float nm = fmaxf(rm[r], lm[r]);
      sc[r] = __expf(rm[r] - nm);
      rm[r] = nm;
      p0[r] = __expf(S0[r] - nm);
      p1[r] = __expf(S1[r] - nm);
      ps[r] = p0[r] + p1[r];
    }
#pragma unroll
    for (int off = 1; off < 16; off <<= 1)
#pragma unroll
      for (int r = 0; r < 8; ++r) ps[r] += __shfl_xor(ps[r], off, 32);
#pragma unroll
    for (int r = 0; r < 8; ++r) rs[r] = rs[r] * sc[r] + ps[r];
#pragma unroll
    for (int t = 0; t < 4; ++t)
#pragma unroll
      for (int r = 0; r < 8; ++r) ctx[t][r] *= sc[r];

    // stage P (C-layout -> row-major) and V^T into LDS
#pragma unroll
    for (int r = 0; r < 8; ++r) {
      Pl[(rbase + r) * 32 + col]      = (bf16_t)p0[r];
      Pl[(rbase + r) * 32 + col + 16] = (bf16_t)p1[r];
    }
    {
      const bf16_t* vp = Vp + (size_t)(s0 + lane) * ld;
#pragma unroll
      for (int d0 = 0; d0 < 64; d0 += 8) {
        u32x4 raw = *(const u32x4*)(vp + d0);
        const bf16_t* rb = (const bf16_t*)&raw;
#pragma unroll
        for (int j = 0; j < 8; ++j) Vt[(d0 + j) * 32 + lane] = rb[j];
      }
    }
    __syncthreads();
    // ctx += P @ V
    v16bf pa = load_frag_row(Pl, 32, 0, 0, lane);
#pragma unroll
    for (int t = 0; t < 4; ++t) {
      v16bf vb = load_frag_col(Vt, 32, t * 16, 0, lane);
      ctx[t] = wmma_bf16(pa, vb, ctx[t]);
    }
    __syncthreads();
  }
  // write O = ctx / rowsum
#pragma unroll
  for (int t = 0; t < 4; ++t)
#pragma unroll
    for (int r = 0; r < 8; ++r) {
      int l = l0 + rbase + r;
      int d = t * 16 + col;
      O[((size_t)b * LLEN + l) * ld + h * HD + d] = (bf16_t)(ctx[t][r] / rs[r]);
    }
}

// ---------------- prob-attn: sampled measure M = max(qk) - sum(qk)/L -----------
__global__ void k_sample_m(const bf16_t* __restrict__ Q, const bf16_t* __restrict__ Km,
                           const int* __restrict__ idx, float* __restrict__ Mm) {
  int t = blockIdx.x * blockDim.x + threadIdx.x;
  if (t >= BB * NH * LLEN) return;
  int l = t % LLEN;
  int bh = t / LLEN;
  int h = bh % NH, b = bh / NH;
  const bf16_t* q = Q + (((size_t)b * LLEN + l) * NH + h) * HD;
  float qv[HD];
#pragma unroll
  for (int d = 0; d < HD; ++d) qv[d] = (float)q[d];
  float mx = -3.0e38f, sm = 0.f;
  for (int s = 0; s < SKEYS; ++s) {
    int ki = idx[l * SKEYS + s];
    const bf16_t* k = Km + (((size_t)b * LLEN + ki) * NH + h) * HD;
    float a = 0.f;
#pragma unroll
    for (int d = 0; d < HD; ++d) a += qv[d] * (float)k[d];
    mx = fmaxf(mx, a);
    sm += a;
  }
  Mm[t] = mx - sm / (float)LLEN;
}

// ---------------- top-35 selection, one wave per (b,h) -------------------------
__global__ __launch_bounds__(32) void k_topk(const float* __restrict__ Mm,
                                             int* __restrict__ mtop) {
  __shared__ float vals[LLEN];
  int lane = threadIdx.x;
  int bh = blockIdx.x;
  const float* m = Mm + (size_t)bh * LLEN;
  for (int i = lane; i < LLEN; i += 32) vals[i] = m[i];
  __syncthreads();
  for (int u = 0; u < NTOP; ++u) {
    float best = -3.0e38f;
    int bi = 0;
    for (int i = lane; i < LLEN; i += 32)
      if (vals[i] > best) { best = vals[i]; bi = i; }
#pragma unroll
    for (int off = 16; off; off >>= 1) {
      float ob = __shfl_xor(best, off, 32);
      int oi = __shfl_xor(bi, off, 32);
      if (ob > best || (ob == best && oi < bi)) { best = ob; bi = oi; }
    }
    if (lane == 0) { mtop[(size_t)bh * NTOP + u] = bi; vals[bi] = -3.0e38f; }
    __syncthreads();
  }
}

// ---------------- mean of V over sequence ----------------
__global__ void k_vmean(const bf16_t* __restrict__ Vm, float* __restrict__ vmean) {
  int t = blockIdx.x * blockDim.x + threadIdx.x;
  if (t >= BB * NH * HD) return;
  int d = t % HD;
  int rem = t / HD;
  int h = rem % NH, b = rem / NH;
  float a = 0.f;
  for (int l = 0; l < LLEN; ++l)
    a += (float)Vm[(((size_t)b * LLEN + l) * NH + h) * HD + d];
  vmean[t] = a / (float)LLEN;
}

// ---------------- broadcast mean-V into ctx2 ----------------
__global__ void k_ctx_base(const float* __restrict__ vmean, bf16_t* __restrict__ ctx2) {
  int t = blockIdx.x * blockDim.x + threadIdx.x;
  if (t >= MROWS * DMODEL) return;
  int c = t % DMODEL;
  int row = t / DMODEL;
  int h = c / HD, d = c % HD;
  int b = row / LLEN;
  ctx2[t] = (bf16_t)vmean[((size_t)b * NH + h) * HD + d];
}

// ---------------- reduced attention for the 35 top queries ---------------------
__global__ __launch_bounds__(32) void k_prob_ctx(const bf16_t* __restrict__ Q,
                                                 const bf16_t* __restrict__ Km,
                                                 const bf16_t* __restrict__ Vm,
                                                 const int* __restrict__ mtop,
                                                 bf16_t* __restrict__ ctx2) {
  __shared__ float sc[LLEN];
  int lane = threadIdx.x;
  int t = blockIdx.x;  // (b*NH + h)*NTOP + u
  int u = t % NTOP;
  int bh = t / NTOP;
  int h = bh % NH, b = bh / NH;
  int lq = mtop[(size_t)bh * NTOP + u];
  const bf16_t* q = Q + (((size_t)b * LLEN + lq) * NH + h) * HD;
  float qv[HD];
#pragma unroll
  for (int d = 0; d < HD; ++d) qv[d] = (float)q[d];
  float mx = -3.0e38f;
  for (int i = lane; i < LLEN; i += 32) {
    const bf16_t* k = Km + (((size_t)b * LLEN + i) * NH + h) * HD;
    float a = 0.f;
#pragma unroll
    for (int d = 0; d < HD; ++d) a += qv[d] * (float)k[d];
    a *= 0.125f;
    sc[i] = a;
    mx = fmaxf(mx, a);
  }
#pragma unroll
  for (int off = 16; off; off >>= 1) mx = fmaxf(mx, __shfl_xor(mx, off, 32));
  float sm = 0.f;
  for (int i = lane; i < LLEN; i += 32) {
    float p = __expf(sc[i] - mx);
    sc[i] = p;
    sm += p;
  }
#pragma unroll
  for (int off = 16; off; off >>= 1) sm += __shfl_xor(sm, off, 32);
  __syncthreads();
  float c0 = 0.f, c1 = 0.f;
  for (int i = 0; i < LLEN; ++i) {
    float p = sc[i];
    const bf16_t* v = Vm + (((size_t)b * LLEN + i) * NH + h) * HD;
    c0 += p * (float)v[lane];
    c1 += p * (float)v[lane + 32];
  }
  bf16_t* o = ctx2 + (((size_t)b * LLEN + lq) * NH + h) * HD;
  o[lane] = (bf16_t)(c0 / sm);
  o[lane + 32] = (bf16_t)(c1 / sm);
}

// ---------------- block-wide sum (256 threads = 8 waves, wave32) ---------------
__device__ __forceinline__ float blk_sum(float v, float* sred) {
#pragma unroll
  for (int off = 16; off; off >>= 1) v += __shfl_xor(v, off, 32);
  int lane = threadIdx.x & 31, wv = threadIdx.x >> 5;
  if (lane == 0) sred[wv] = v;
  __syncthreads();
  float s = (threadIdx.x < 8) ? sred[threadIdx.x] : 0.f;
#pragma unroll
  for (int off = 4; off; off >>= 1) s += __shfl_xor(s, off, 32);
  if (threadIdx.x == 0) sred[0] = s;
  __syncthreads();
  float r = sred[0];
  __syncthreads();
  return r;
}

// ---------------- gated residual + LayerNorm1 ----------------
__global__ __launch_bounds__(256) void k_gate_ln(float* __restrict__ hf,
                                                 const float* __restrict__ a1,
                                                 const float* __restrict__ a2,
                                                 const float* __restrict__ gate,
                                                 int layer, const float* __restrict__ g,
                                                 const float* __restrict__ bta,
                                                 bf16_t* __restrict__ hb) {
  __shared__ float sred[8];
  int row = blockIdx.x, tid = threadIdx.x;
  float gv = 1.0f / (1.0f + __expf(-gate[layer]));
  size_t base = (size_t)row * DMODEL;
  float x[2];
  float s = 0.f;
#pragma unroll
  for (int j = 0; j < 2; ++j) {
    int c = tid + j * 256;
    float v = hf[base + c] + gv * a1[base + c] + (1.0f - gv) * a2[base + c];
    x[j] = v;
    s += v;
  }
  float mean = blk_sum(s, sred) / (float)DMODEL;
  float s2 = 0.f;
#pragma unroll
  for (int j = 0; j < 2; ++j) { float d = x[j] - mean; s2 += d * d; }
  float var = blk_sum(s2, sred) / (float)DMODEL;
  float rstd = rsqrtf(var + LN_EPS);
#pragma unroll
  for (int j = 0; j < 2; ++j) {
    int c = tid + j * 256;
    float y = (x[j] - mean) * rstd * g[c] + bta[c];
    hf[base + c] = y;
    hb[base + c] = (bf16_t)y;
  }
}

// ---------------- FFN residual + LayerNorm2 + layer-weight accumulate ----------
__global__ __launch_bounds__(256) void k_ff_ln(float* __restrict__ hf,
                                               const float* __restrict__ ff,
                                               const float* __restrict__ g,
                                               const float* __restrict__ bta,
                                               const float* __restrict__ lw, int layer,
                                               int first, bf16_t* __restrict__ hb,
                                               float* __restrict__ acc) {
  __shared__ float sred[8];
  int row = blockIdx.x, tid = threadIdx.x;
  size_t base = (size_t)row * DMODEL;
  float x[2];
  float s = 0.f;
#pragma unroll
  for (int j = 0; j < 2; ++j) {
    int c = tid + j * 256;
    float v = hf[base + c] + ff[base + c];
    x[j] = v;
    s += v;
  }
  float mean = blk_sum(s, sred) / (float)DMODEL;
  float s2 = 0.f;
#pragma unroll
  for (int j = 0; j < 2; ++j) { float d = x[j] - mean; s2 += d * d; }
  float var = blk_sum(s2, sred) / (float)DMODEL;
  float rstd = rsqrtf(var + LN_EPS);
  // softmax over the 3 layer weights
  float w0 = lw[0], w1 = lw[1], w2 = lw[2];
  float wm = fmaxf(w0, fmaxf(w1, w2));
  float e0 = __expf(w0 - wm), e1 = __expf(w1 - wm), e2 = __expf(w2 - wm);
  float esum = e0 + e1 + e2;
  float nw = ((layer == 0) ? e0 : (layer == 1) ? e1 : e2) / esum;
#pragma unroll
  for (int j = 0; j < 2; ++j) {
    int c = tid + j * 256;
    float y = (x[j] - mean) * rstd * g[c] + bta[c];
    hf[base + c] = y;
    hb[base + c] = (bf16_t)y;
    if (first) acc[base + c] = nw * y;
    else       acc[base + c] += nw * y;
  }
}

// ---------------- final projection to OD=1 ----------------
__global__ __launch_bounds__(256) void k_out(const float* __restrict__ acc,
                                             const float* __restrict__ ow,
                                             const float* __restrict__ ob,
                                             float* __restrict__ out) {
  __shared__ float sred[8];
  int row = blockIdx.x, tid = threadIdx.x;
  size_t base = (size_t)row * DMODEL;
  float s = 0.f;
#pragma unroll
  for (int j = 0; j < 2; ++j) {
    int c = tid + j * 256;
    s += acc[base + c] * ow[c];
  }
  float tot = blk_sum(s, sred);
  if (tid == 0) out[row] = tot + ob[0];
}

// =====================================================================
extern "C" void kernel_launch(void* const* d_in, const int* in_sizes, int n_in,
                              void* d_out, int out_size, void* d_ws, size_t ws_size,
                              hipStream_t stream) {
  (void)in_sizes; (void)n_in; (void)out_size; (void)ws_size;
  const float* x      = (const float*)d_in[0];
  const int*   sidx   = (const int*)d_in[1];
  const float* emb_w  = (const float*)d_in[2];
  const float* emb_b  = (const float*)d_in[3];
  const float* pe     = (const float*)d_in[4];
  const float* pe_w   = (const float*)d_in[5];
  const float* ff1_w  = (const float*)d_in[6];
  const float* ff1_b  = (const float*)d_in[7];
  const float* ff2_w  = (const float*)d_in[8];
  const float* ff2_b  = (const float*)d_in[9];
  const float* ln1_g  = (const float*)d_in[10];
  const float* ln1_b  = (const float*)d_in[11];
  const float* ln2_g  = (const float*)d_in[12];
  const float* ln2_b  = (const float*)d_in[13];
  const float* gate   = (const float*)d_in[14];
  const float* layerw = (const float*)d_in[15];
  const float* out_w  = (const float*)d_in[16];
  const float* out_b  = (const float*)d_in[17];
  const float* a1_qw  = (const float*)d_in[18];
  const float* a1_kw  = (const float*)d_in[19];
  const float* a1_vw  = (const float*)d_in[20];
  const float* a1_ow  = (const float*)d_in[21];
  const float* a1_qb  = (const float*)d_in[22];
  const float* a1_kb  = (const float*)d_in[23];
  const float* a1_vb  = (const float*)d_in[24];
  const float* a1_ob  = (const float*)d_in[25];
  const float* a2_qw  = (const float*)d_in[26];
  const float* a2_kw  = (const float*)d_in[27];
  const float* a2_vw  = (const float*)d_in[28];
  const float* a2_ow  = (const float*)d_in[29];
  const float* a2_qb  = (const float*)d_in[30];
  const float* a2_kb  = (const float*)d_in[31];
  const float* a2_vb  = (const float*)d_in[32];
  const float* a2_ob  = (const float*)d_in[33];
  float* out = (float*)d_out;

  // ---- workspace bump allocator ----
  char* arena = (char*)d_ws;
  size_t off = 0;
  auto take = [&](size_t bytes) -> void* {
    void* p = arena + off;
    off = (off + bytes + 255) & ~(size_t)255;
    return p;
  };
  float*  hf   = (float*) take((size_t)MROWS * DMODEL * 4);
  bf16_t* hb   = (bf16_t*)take((size_t)MROWS * DMODEL * 2);
  bf16_t* qb   = (bf16_t*)take((size_t)MROWS * DMODEL * 2);
  bf16_t* kb   = (bf16_t*)take((size_t)MROWS * DMODEL * 2);
  bf16_t* vb   = (bf16_t*)take((size_t)MROWS * DMODEL * 2);
  bf16_t* cb   = (bf16_t*)take((size_t)MROWS * DMODEL * 2);  // ctx (shared a1/a2)
  float*  a1f  = (float*) take((size_t)MROWS * DMODEL * 4);
  float*  a2f  = (float*) take((size_t)MROWS * DMODEL * 4);
  bf16_t* ffb  = (bf16_t*)take((size_t)MROWS * DFFN * 2);
  float*  fff  = (float*) take((size_t)MROWS * DMODEL * 4);
  float*  accf = (float*) take((size_t)MROWS * DMODEL * 4);
  float*  Mm   = (float*) take((size_t)BB * NH * LLEN * 4);
  int*    mtop = (int*)   take((size_t)BB * NH * NTOP * 4);
  float*  vmn  = (float*) take((size_t)BB * NH * HD * 4);
  bf16_t* wq1  = (bf16_t*)take((size_t)DMODEL * DMODEL * 2);
  bf16_t* wk1  = (bf16_t*)take((size_t)DMODEL * DMODEL * 2);
  bf16_t* wv1  = (bf16_t*)take((size_t)DMODEL * DMODEL * 2);
  bf16_t* wo1  = (bf16_t*)take((size_t)DMODEL * DMODEL * 2);
  bf16_t* wq2  = (bf16_t*)take((size_t)DMODEL * DMODEL * 2);
  bf16_t* wk2  = (bf16_t*)take((size_t)DMODEL * DMODEL * 2);
  bf16_t* wv2  = (bf16_t*)take((size_t)DMODEL * DMODEL * 2);
  bf16_t* wo2  = (bf16_t*)take((size_t)DMODEL * DMODEL * 2);
  bf16_t* wf1  = (bf16_t*)take((size_t)DMODEL * DFFN * 2);
  bf16_t* wf2  = (bf16_t*)take((size_t)DFFN * DMODEL * 2);

  const size_t WSQ = (size_t)DMODEL * DMODEL;

  // embedding
  k_embed<<<(MROWS * DMODEL + 255) / 256, 256, 0, stream>>>(x, emb_w, emb_b, pe,
                                                            pe_w, hf, hb);

  dim3 gQ(MROWS / 256, DMODEL / 16);  // (32, 32)
  dim3 gF1(MROWS / 256, DFFN / 16);   // (32, 128)
  dim3 gAtt(LLEN / 16, BB * NH);      // (64, 64)

  for (int layer = 0; layer < NLAYERS; ++layer) {
    // --- convert weights (f32 -> bf16, transposed to (N,K)) ---
    int nwb = (int)((WSQ + 255) / 256);
    k_wt<<<nwb, 256, 0, stream>>>(a1_qw + layer * WSQ, wq1, DMODEL, DMODEL);
    k_wt<<<nwb, 256, 0, stream>>>(a1_kw + layer * WSQ, wk1, DMODEL, DMODEL);
    k_wt<<<nwb, 256, 0, stream>>>(a1_vw + layer * WSQ, wv1, DMODEL, DMODEL);
    k_wt<<<nwb, 256, 0, stream>>>(a1_ow + layer * WSQ, wo1, DMODEL, DMODEL);
    k_wt<<<nwb, 256, 0, stream>>>(a2_qw + layer * WSQ, wq2, DMODEL, DMODEL);
    k_wt<<<nwb, 256, 0, stream>>>(a2_kw + layer * WSQ, wk2, DMODEL, DMODEL);
    k_wt<<<nwb, 256, 0, stream>>>(a2_vw + layer * WSQ, wv2, DMODEL, DMODEL);
    k_wt<<<nwb, 256, 0, stream>>>(a2_ow + layer * WSQ, wo2, DMODEL, DMODEL);
    int nwf = (int)(((size_t)DMODEL * DFFN + 255) / 256);
    k_wt<<<nwf, 256, 0, stream>>>(ff1_w + (size_t)layer * DMODEL * DFFN, wf1,
                                  DMODEL, DFFN);
    k_wt<<<nwf, 256, 0, stream>>>(ff2_w + (size_t)layer * DFFN * DMODEL, wf2,
                                  DFFN, DMODEL);

    // --- attention branch 1 (full attention) ---
    k_gemm<<<gQ, 128, 0, stream>>>(hb, wq1, a1_qb + layer * DMODEL, nullptr, qb,
                                   DMODEL, DMODEL, 0);
    k_gemm<<<gQ, 128, 0, stream>>>(hb, wk1, a1_kb + layer * DMODEL, nullptr, kb,
                                   DMODEL, DMODEL, 0);
    k_gemm<<<gQ, 128, 0, stream>>>(hb, wv1, a1_vb + layer * DMODEL, nullptr, vb,
                                   DMODEL, DMODEL, 0);
    k_flash<<<gAtt, 32, 0, stream>>>(qb, kb, vb, cb);
    k_gemm<<<gQ, 128, 0, stream>>>(cb, wo1, a1_ob + layer * DMODEL, a1f, nullptr,
                                   DMODEL, DMODEL, 0);

    // --- attention branch 2 (prob-sparse attention) ---
    k_gemm<<<gQ, 128, 0, stream>>>(hb, wq2, a2_qb + layer * DMODEL, nullptr, qb,
                                   DMODEL, DMODEL, 0);
    k_gemm<<<gQ, 128, 0, stream>>>(hb, wk2, a2_kb + layer * DMODEL, nullptr, kb,
                                   DMODEL, DMODEL, 0);
    k_gemm<<<gQ, 128, 0, stream>>>(hb, wv2, a2_vb + layer * DMODEL, nullptr, vb,
                                   DMODEL, DMODEL, 0);
    k_sample_m<<<(BB * NH * LLEN + 255) / 256, 256, 0, stream>>>(
        qb, kb, sidx + (size_t)layer * LLEN * SKEYS, Mm);
    k_topk<<<BB * NH, 32, 0, stream>>>(Mm, mtop);
    k_vmean<<<(BB * NH * HD + 255) / 256, 256, 0, stream>>>(vb, vmn);
    k_ctx_base<<<(MROWS * DMODEL + 255) / 256, 256, 0, stream>>>(vmn, cb);
    k_prob_ctx<<<BB * NH * NTOP, 32, 0, stream>>>(qb, kb, vb, mtop, cb);
    k_gemm<<<gQ, 128, 0, stream>>>(cb, wo2, a2_ob + layer * DMODEL, a2f, nullptr,
                                   DMODEL, DMODEL, 0);

    // --- gated residual + LN1 ---
    k_gate_ln<<<MROWS, 256, 0, stream>>>(hf, a1f, a2f, gate, layer,
                                         ln1_g + layer * DMODEL,
                                         ln1_b + layer * DMODEL, hb);

    // --- FFN with exact GELU ---
    k_gemm<<<gF1, 128, 0, stream>>>(hb, wf1, ff1_b + layer * DFFN, nullptr, ffb,
                                    DFFN, DMODEL, 1);
    k_gemm<<<gQ, 128, 0, stream>>>(ffb, wf2, ff2_b + layer * DMODEL, fff, nullptr,
                                   DMODEL, DFFN, 0);

    // --- residual + LN2 + weighted layer accumulation ---
    k_ff_ln<<<MROWS, 256, 0, stream>>>(hf, fff, ln2_g + layer * DMODEL,
                                       ln2_b + layer * DMODEL, layerw, layer,
                                       layer == 0 ? 1 : 0, hb, accf);
  }

  // final projection (DM -> 1)
  k_out<<<MROWS, 256, 0, stream>>>(accf, out_w, out_b, out);
}
